// Loss_19980187861563
// MI455X (gfx1250) — matrix operand order repaired and checked by reading, not compiled, not run on hardware
//
#include <hip/hip_runtime.h>

#define BB 4
#define TT 128
#define VV 50257
#define LSRCN 512
#define DD 512
#define NN 512   /* B*T */

#define BM 128
#define BN 128
#define BK 64
#define KSPLIT 16
#define KCHUNK 3200          /* per-split K range, multiple of BK; 16*3200 >= V */
#define LDA (BK + 8)         /* 72: rows 16B-aligned -> ds_load_b128 fragments */
#define LDBK (BK + 8)        /* B tile stored transposed [BN][BK] */
#define A_BUF (BM * LDA)     /* 9216 ushorts per buffer */
#define B_BUF (BN * LDBK)

typedef __attribute__((ext_vector_type(16))) __bf16 v16bf;
typedef __attribute__((ext_vector_type(8)))  float  v8f;

static __device__ __forceinline__ unsigned short f2bf(float f) {
  unsigned int u = __float_as_uint(f);
  u += 0x7FFFu + ((u >> 16) & 1u);          // round-to-nearest-even
  return (unsigned short)(u >> 16);
}

static __device__ __forceinline__ unsigned int pack2bf(float x0, float x1) {
#if __has_builtin(__builtin_amdgcn_cvt_pk_bf16_f32)
  typedef __attribute__((ext_vector_type(2))) __bf16 v2bf;
  v2bf v = __builtin_amdgcn_cvt_pk_bf16_f32(x0, x1);
  return __builtin_bit_cast(unsigned int, v);
#else
  return (unsigned int)f2bf(x0) | ((unsigned int)f2bf(x1) << 16);
#endif
}

// ---------------- Kernel A: NLL + coverage partial sums -----------------
__global__ void nll_cov_kernel(const float* __restrict__ outp,
                               const float* __restrict__ attn,
                               const float* __restrict__ covr,
                               const int* __restrict__ trg,
                               const unsigned char* __restrict__ dmask,
                               float* __restrict__ partA) {
  const int tid = threadIdx.x;
  const int gid = blockIdx.x * 256 + tid;
  const int gsz = gridDim.x * 256;
  float nsum = 0.f, ncnt = 0.f, csum = 0.f;
  for (int i = gid; i < NN; i += gsz) {
    int t = trg[i];
    if (t != 0) {                                    // PAD_ID == 0
      nsum += __logf(outp[(size_t)i * VV + t]);
      ncnt += 1.f;
    }
    if (!dmask[i]) {
      int b = i >> 7, tc = i & (TT - 1);
      const float* ap = attn + (size_t)b * LSRCN * TT + tc;
      const float* cp = covr + (size_t)b * LSRCN * TT + tc;
      float s = 0.f;
      for (int l = 0; l < LSRCN; ++l)
        s += fminf(ap[(size_t)l * TT], cp[(size_t)l * TT]);
      csum += s;
    }
  }
  __shared__ float s0[256], s1[256], s2[256];
  s0[tid] = nsum; s1[tid] = ncnt; s2[tid] = csum;
  __syncthreads();
  for (int off = 128; off > 0; off >>= 1) {
    if (tid < off) { s0[tid] += s0[tid+off]; s1[tid] += s1[tid+off]; s2[tid] += s2[tid+off]; }
    __syncthreads();
  }
  if (tid == 0) {
    partA[blockIdx.x * 3 + 0] = s0[0];
    partA[blockIdx.x * 3 + 1] = s1[0];
    partA[blockIdx.x * 3 + 2] = s2[0];
  }
}

// ---------------- Kernel B: per-row online softmax stats -----------------
__global__ void softmax_stats_kernel(const float* __restrict__ outp,
                                     float* __restrict__ rmax,
                                     float* __restrict__ rsum) {
  const int row = blockIdx.x;
  const int tid = threadIdx.x;
  const float* p = outp + (size_t)row * VV;
  float m = -3.4e38f, s = 0.f;
  for (int k = tid; k < VV; k += 256) {
    float x = p[k];
    if (x > m) { s = s * __expf(m - x) + 1.f; m = x; }
    else       { s += __expf(x - m); }
  }
  __shared__ float sm[256], ss[256];
  sm[tid] = m; ss[tid] = s;
  __syncthreads();
  for (int off = 128; off > 0; off >>= 1) {
    if (tid < off) {
      float m2 = sm[tid+off], s2 = ss[tid+off];
      float M  = fmaxf(sm[tid], m2);
      ss[tid]  = ss[tid] * __expf(sm[tid] - M) + s2 * __expf(m2 - M);
      sm[tid]  = M;
    }
    __syncthreads();
  }
  if (tid == 0) { rmax[row] = sm[0]; rsum[row] = ss[0]; }
}

// ---------- Kernel C: double-buffered bf16 WMMA GEMM --------------------
// predp[split] = exp(outp - rowmax) @ W_emb  (unnormalized probs x embeddings)
__global__ void __launch_bounds__(256)
gemm_probs_wemb_kernel(const float* __restrict__ outp,
                       const float* __restrict__ wemb,
                       const float* __restrict__ rmax,
                       float* __restrict__ predp) {
  extern __shared__ unsigned short smem[];
  unsigned short* const As0 = smem;
  unsigned short* const As1 = smem + A_BUF;
  unsigned short* const Bs0 = smem + 2 * A_BUF;
  unsigned short* const Bs1 = smem + 2 * A_BUF + B_BUF;

  const int tid = threadIdx.x;
  const int rowBase = blockIdx.y * BM;
  const int colBase = blockIdx.x * BN;
  const int kBeg = blockIdx.z * KCHUNK;
  int kEnd = kBeg + KCHUNK; if (kEnd > VV) kEnd = VV;

  const int lane = tid & 31;
  const int wave = tid >> 5;
  const int wr = wave >> 1;            // 0..3 : 32-row slab
  const int wc = wave & 1;             // 0..1 : 64-col slab
  const int hi = (lane >> 4) & 1;
  const int ln = lane & 15;

  // Row maxes for the 16 A-rows this thread stages (wave-uniform addresses).
  float rmaxR[16];
  #pragma unroll
  for (int i = 0; i < 16; ++i) rmaxR[i] = rmax[rowBase + (tid >> 5) + i * 8];

  float  aReg[32];
  float4 bReg[8];

  // A staging: pair pi = tid + i*256 (i<16): row r = pi>>5, col pair cp = pi&31
  // B staging: pair pi = tid + i*256 (i<4):  row pair rp = pi>>5, col4 c4 = pi&31
  auto loadRegs = [&](int k0) {
    if (k0 + BK <= VV) {               // fast path: no bounds checks
      #pragma unroll
      for (int i = 0; i < 16; ++i) {
        int pi = tid + i * 256;
        int r = pi >> 5, cp = pi & 31;
        const float* src = outp + (size_t)(rowBase + r) * VV + k0 + 2 * cp;
        aReg[2*i]   = src[0];
        aReg[2*i+1] = src[1];
      }
      #pragma unroll
      for (int i = 0; i < 4; ++i) {
        int pi = tid + i * 256;
        int rp = pi >> 5, c4 = pi & 31;
        const float* src = wemb + (size_t)(k0 + 2 * rp) * DD + colBase + c4 * 4;
        bReg[2*i]   = *(const float4*)(src);
        bReg[2*i+1] = *(const float4*)(src + DD);
      }
    } else {                           // guarded tail (last stage of last split)
      #pragma unroll
      for (int i = 0; i < 16; ++i) {
        int pi = tid + i * 256;
        int r = pi >> 5, cp = pi & 31;
        int kg = k0 + 2 * cp;
        const float* src = outp + (size_t)(rowBase + r) * VV + kg;
        aReg[2*i]   = (kg     < VV) ? src[0] : 0.f;
        aReg[2*i+1] = (kg + 1 < VV) ? src[1] : 0.f;
      }
      #pragma unroll
      for (int i = 0; i < 4; ++i) {
        int pi = tid + i * 256;
        int rp = pi >> 5, c4 = pi & 31;
        int kg = k0 + 2 * rp;
        const float* src = wemb + (size_t)kg * DD + colBase + c4 * 4;
        bReg[2*i]   = (kg     < VV) ? *(const float4*)(src)      : make_float4(0.f,0.f,0.f,0.f);
        bReg[2*i+1] = (kg + 1 < VV) ? *(const float4*)(src + DD) : make_float4(0.f,0.f,0.f,0.f);
      }
    }
  };

  auto storeLDS = [&](unsigned short* Ab, unsigned short* Bb, int k0) {
    if (k0 + BK <= VV) {
      #pragma unroll
      for (int i = 0; i < 16; ++i) {
        int pi = tid + i * 256;
        int r = pi >> 5, cp = pi & 31;
        float x0 = __expf(aReg[2*i]   - rmaxR[i]);
        float x1 = __expf(aReg[2*i+1] - rmaxR[i]);
        *(unsigned int*)&Ab[r * LDA + 2 * cp] = pack2bf(x0, x1);
      }
    } else {
      #pragma unroll
      for (int i = 0; i < 16; ++i) {
        int pi = tid + i * 256;
        int r = pi >> 5, cp = pi & 31;
        int kg = k0 + 2 * cp;
        float x0 = (kg     < VV) ? __expf(aReg[2*i]   - rmaxR[i]) : 0.f;
        float x1 = (kg + 1 < VV) ? __expf(aReg[2*i+1] - rmaxR[i]) : 0.f;
        *(unsigned int*)&Ab[r * LDA + 2 * cp] = pack2bf(x0, x1);
      }
    }
    #pragma unroll
    for (int i = 0; i < 4; ++i) {      // B: zeros already loaded for OOB rows
      int pi = tid + i * 256;
      int rp = pi >> 5, c4 = pi & 31;
      unsigned int* bp = (unsigned int*)&Bb[(c4 * 4) * LDBK + 2 * rp];
      float4 lo = bReg[2*i], hv = bReg[2*i+1];
      bp[0 * (LDBK/2)] = pack2bf(lo.x, hv.x);
      bp[1 * (LDBK/2)] = pack2bf(lo.y, hv.y);
      bp[2 * (LDBK/2)] = pack2bf(lo.z, hv.z);
      bp[3 * (LDBK/2)] = pack2bf(lo.w, hv.w);
    }
  };

  v8f acc[2][4] = {};

  auto compute = [&](const unsigned short* Ab, const unsigned short* Bb) {
    #pragma unroll
    for (int ks = 0; ks < BK; ks += 32) {
      // A frag (16x32 bf16): ushorts [0..7] and [16..23] of the row slice
      union { v16bf v; uint4 q[2]; } afr[2], bfr[4];
      #pragma unroll
      for (int m = 0; m < 2; ++m) {
        const unsigned short* ap = &Ab[(wr*32 + m*16 + ln) * LDA + ks + hi*8];
        afr[m].q[0] = *(const uint4*)(ap);
        afr[m].q[1] = *(const uint4*)(ap + 16);
      }
      // B frag (32x16 bf16): ushorts [0..7] and [16..23] of the column slice
      #pragma unroll
      for (int n = 0; n < 4; ++n) {
        const unsigned short* bp = &Bb[(wc*64 + n*16 + ln) * LDBK + ks + hi*8];
        bfr[n].q[0] = *(const uint4*)(bp);
        bfr[n].q[1] = *(const uint4*)(bp + 16);
      }
      #pragma unroll
      for (int m = 0; m < 2; ++m)
        #pragma unroll
        for (int n = 0; n < 4; ++n)
          acc[m][n] = __builtin_amdgcn_wmma_f32_16x16x32_bf16(
              false, afr[m].v, false, bfr[n].v, (short)0, acc[m][n], false, false);
    }
  };

  // prologue: fill buffer 0
  loadRegs(kBeg);
  storeLDS(As0, Bs0, kBeg);
  __syncthreads();

  int k0 = kBeg;
  for (;;) {
    int kn = k0 + BK;
    bool more = kn < kEnd;
    if (more) loadRegs(kn);            // overlap next-stage HBM reads with WMMAs
    compute(As0, Bs0);
    if (more) storeLDS(As1, Bs1, kn);
    __syncthreads();
    k0 = kn;
    if (!more) break;

    kn = k0 + BK;
    more = kn < kEnd;
    if (more) loadRegs(kn);
    compute(As1, Bs1);
    if (more) storeLDS(As0, Bs0, kn);
    __syncthreads();
    k0 = kn;
    if (!more) break;
  }

  float* pp = predp + (size_t)blockIdx.z * NN * DD;
  #pragma unroll
  for (int m = 0; m < 2; ++m)
    #pragma unroll
    for (int n = 0; n < 4; ++n) {
      int col = colBase + wc*64 + n*16 + ln;
      #pragma unroll
      for (int r = 0; r < 8; ++r) {
        int row = rowBase + wr*32 + m*16 + hi*8 + r;
        pp[(size_t)row * DD + col] = acc[m][n][r];
      }
    }
}

// -------- Kernel C2: sum split-K partials, scale by 1/rowsum ------------
__global__ void reduce_pred_kernel(const float* __restrict__ predp,
                                   const float* __restrict__ rsum,
                                   float* __restrict__ pred) {
  const int idx = blockIdx.x * 256 + threadIdx.x;   // 0 .. N*D-1
  const int row = idx >> 9;                          // / DD
  float s = 0.f;
  #pragma unroll
  for (int sp = 0; sp < KSPLIT; ++sp)
    s += predp[(size_t)sp * NN * DD + idx];
  pred[idx] = s / rsum[row];
}

// -------- Kernel D: per-row cosine(pred_i, W_emb[trg_i]) ----------------
__global__ void cosine_kernel(const float* __restrict__ pred,
                              const float* __restrict__ wemb,
                              const int* __restrict__ trg,
                              float* __restrict__ cosv) {
  const int row = blockIdx.x;
  const int tid = threadIdx.x;
  const float* p = pred + (size_t)row * DD;
  const float* e = wemb + (size_t)trg[row] * DD;
  float d = 0.f, np = 0.f, ne = 0.f;
  for (int k = tid; k < DD; k += 256) {
    float a = p[k], b = e[k];
    d += a * b; np += a * a; ne += b * b;
  }
  __shared__ float s0[256], s1[256], s2[256];
  s0[tid] = d; s1[tid] = np; s2[tid] = ne;
  __syncthreads();
  for (int off = 128; off > 0; off >>= 1) {
    if (tid < off) { s0[tid] += s0[tid+off]; s1[tid] += s1[tid+off]; s2[tid] += s2[tid+off]; }
    __syncthreads();
  }
  if (tid == 0) cosv[row] = s0[0] * rsqrtf(s1[0] * s2[0]);
}

// -------- Kernel E: final scalar combine --------------------------------
__global__ void finalize_kernel(const float* __restrict__ cosv,
                                const float* __restrict__ partA,
                                const int* __restrict__ dec_len,
                                float* __restrict__ out) {
  const int tid = threadIdx.x;
  float c = 0.f;
  for (int i = tid; i < NN; i += 256) c += cosv[i];
  __shared__ float s0[256];
  s0[tid] = c;
  __syncthreads();
  for (int off = 128; off > 0; off >>= 1) {
    if (tid < off) s0[tid] += s0[tid+off];
    __syncthreads();
  }
  if (tid == 0) {
    float nsum = 0.f, ncnt = 0.f, csum = 0.f;
    for (int b = 0; b < 64; ++b) {
      nsum += partA[b*3 + 0];
      ncnt += partA[b*3 + 1];
      csum += partA[b*3 + 2];
    }
    float dl = 0.f;
    for (int b = 0; b < BB; ++b) dl += (float)dec_len[b];
    float nll = -nsum / ncnt;
    float cov_loss = csum / dl;
    float ot = s0[0] / (float)NN;   // IPOT fixed point: mean diag cosine
    out[0] = nll + 1.0f * cov_loss + 0.1f + ot;
  }
}

extern "C" void kernel_launch(void* const* d_in, const int* in_sizes, int n_in,
                              void* d_out, int out_size, void* d_ws, size_t ws_size,
                              hipStream_t stream) {
  const float* outp = (const float*)d_in[0];          // [512, 50257]
  const float* attn = (const float*)d_in[1];          // [4, 512, 128]
  const float* covr = (const float*)d_in[2];          // [4, 512, 128]
  const int*   trg  = (const int*)d_in[3];            // [512]
  const unsigned char* dmask = (const unsigned char*)d_in[4]; // [512] bool
  const int* dec_len = (const int*)d_in[5];           // [4]
  const float* wemb  = (const float*)d_in[6];         // [50257, 512]
  float* out = (float*)d_out;

  float* wsf   = (float*)d_ws;
  float* rmax  = wsf;                         // 512
  float* rsum  = wsf + 512;                   // 512
  float* cosv  = wsf + 1024;                  // 512
  float* partA = wsf + 1536;                  // 64*3
  float* pred  = wsf + 2048;                  // 512*512
  float* predp = wsf + 2048 + NN * DD;        // 16 * 512*512 (~17 MB total)

  nll_cov_kernel<<<64, 256, 0, stream>>>(outp, attn, covr, trg, dmask, partA);
  softmax_stats_kernel<<<NN, 256, 0, stream>>>(outp, rmax, rsum);

  dim3 g(DD / BN, NN / BM, KSPLIT);           // 4 x 4 x 16 = 256 WGs
  size_t shmem = (size_t)(2 * A_BUF + 2 * B_BUF) * sizeof(unsigned short); // 73,728 B
  gemm_probs_wemb_kernel<<<g, 256, shmem, stream>>>(outp, wemb, rmax, predp);

  reduce_pred_kernel<<<(NN * DD) / 256, 256, 0, stream>>>(predp, rsum, pred);
  cosine_kernel<<<NN, 256, 0, stream>>>(pred, wemb, trg, cosv);
  finalize_kernel<<<1, 256, 0, stream>>>(cosv, partA, dec_len, out);
}
